// MixtureOfDepth_17360257810706
// MI455X (gfx1250) — compile-verified, matrix operand order checked by reading, not run on hardware
//
#include <hip/hip_runtime.h>
#include <hip/hip_bf16.h>

#define B_   4
#define S_   8192
#define D_   1024
#define CAP_ 4096
#define DFF_ 4096
#define M_   (B_ * CAP_)   // 16384 selected rows total

typedef __attribute__((ext_vector_type(16))) __bf16 v16bf;
typedef __attribute__((ext_vector_type(8)))  __bf16 v8bf;
typedef __attribute__((ext_vector_type(8)))  float  v8f;
typedef int v4i __attribute__((vector_size(16)));

#if defined(__gfx1250__) && __has_builtin(__builtin_amdgcn_global_load_async_to_lds_b128) && \
    __has_builtin(__builtin_amdgcn_s_wait_asynccnt)
#define USE_ASYNC_LDS 1
#else
#define USE_ASYNC_LDS 0
#endif

// ---------- helpers ----------
static __device__ inline unsigned short f2bf(float f) {
    unsigned int u = __float_as_uint(f);
    unsigned int r = u + 0x7FFFu + ((u >> 16) & 1u);   // round-to-nearest-even
    return (unsigned short)(r >> 16);
}
static __device__ inline unsigned int pack2bf(float a, float b) {
    return (unsigned int)f2bf(a) | ((unsigned int)f2bf(b) << 16);
}

#if USE_ASYNC_LDS
static __device__ inline void async_copy16(const unsigned short* g, unsigned short* l) {
    __builtin_amdgcn_global_load_async_to_lds_b128(
        (__attribute__((address_space(1))) v4i*)g,
        (__attribute__((address_space(3))) v4i*)l, 0, 0);
}
#endif

// ---------- 1) router logits: one wave per token, shuffle reduction ----------
__global__ __launch_bounds__(256) void router_kernel(const float* __restrict__ x,
                                                     const float* __restrict__ Wg,
                                                     const float* __restrict__ bg,
                                                     float* __restrict__ logits) {
    int gwave = (blockIdx.x * blockDim.x + threadIdx.x) >> 5;  // token id
    int lane  = threadIdx.x & 31;
    const float* xr = x + (size_t)gwave * D_;
    float s = 0.f;
    #pragma unroll 8
    for (int k = lane; k < D_; k += 32) s += xr[k] * Wg[k];
    #pragma unroll
    for (int off = 16; off > 0; off >>= 1) s += __shfl_xor(s, off, 32);
    if (lane == 0) logits[gwave] = s + bg[0];
}

// ---------- 2) per-batch bitonic top-CAP (softmax is monotonic -> sort logits) ----------
__global__ __launch_bounds__(1024) void topk_kernel(const float* __restrict__ logits,
                                                    int* __restrict__ idx_out) {
    __shared__ float key[S_];
    __shared__ unsigned short val[S_];   // indices fit in 16 bits (S=8192)
    const float* lg = logits + (size_t)blockIdx.x * S_;
    for (int i = threadIdx.x; i < S_; i += 1024) { key[i] = lg[i]; val[i] = (unsigned short)i; }
    __syncthreads();
    for (int k = 2; k <= S_; k <<= 1) {
        for (int j = k >> 1; j > 0; j >>= 1) {
            for (int t = threadIdx.x; t < S_; t += 1024) {
                int ixj = t ^ j;
                if (ixj > t) {
                    bool ddir = ((t & k) == 0);          // descending segments
                    float a = key[t], c = key[ixj];
                    if ((a < c) == ddir) {
                        key[t] = c; key[ixj] = a;
                        unsigned short v = val[t]; val[t] = val[ixj]; val[ixj] = v;
                    }
                }
            }
            __syncthreads();
        }
    }
    for (int i = threadIdx.x; i < CAP_; i += 1024)
        idx_out[(size_t)blockIdx.x * CAP_ + i] = (int)val[i];
}

// ---------- 3) gather selected tokens -> bf16 (vectorized f32x4 -> bf16x4) ----------
__global__ __launch_bounds__(256) void gather_kernel(const float* __restrict__ x,
                                                     const int* __restrict__ idx,
                                                     unsigned int* __restrict__ selA) {
    int row = blockIdx.x;                 // 0 .. M_-1
    int b   = row >> 12;                  // / CAP_
    int tok = idx[row];
    const float4* src = (const float4*)(x + ((size_t)b * S_ + tok) * D_);
    unsigned int* dst = selA + (size_t)row * (D_ / 2);
    int t = threadIdx.x;                  // 256 threads x 4 floats = 1024 = D_
    float4 v = src[t];
    dst[t * 2 + 0] = pack2bf(v.x, v.y);
    dst[t * 2 + 1] = pack2bf(v.z, v.w);
}

// ---------- 3b) f32 -> bf16 weight convert (4 elems / thread) ----------
__global__ __launch_bounds__(256) void conv_bf16_kernel(const float4* __restrict__ w,
                                                        unsigned int* __restrict__ o, int n4) {
    int i = blockIdx.x * 256 + threadIdx.x;
    if (i < n4) {
        float4 v = w[i];
        o[i * 2 + 0] = pack2bf(v.x, v.y);
        o[i * 2 + 1] = pack2bf(v.z, v.w);
    }
}

// ---------- 4) pass-through copy x -> out ----------
__global__ __launch_bounds__(256) void copy_kernel(const float4* __restrict__ x,
                                                   float4* __restrict__ out, int n4) {
    int i = blockIdx.x * 256 + threadIdx.x;
    if (i < n4) out[i] = x[i];
}

// ---------- 5) bf16 WMMA GEMM, 128x64 block tile, 8 waves of 32x32 ----------
// Double-buffered LDS software pipeline: one barrier / K-chunk; next tile's
// global loads (async-to-LDS for A when available) overlap the WMMA work.
// LDS row stride 40 ushorts (= 20 dwords): lanes ml=0..15 hit banks
// {0,4,...,60}; 4-bank-wide b128 accesses tile all 64 banks -> conflict-free.
// mode 0: H = bf16( gelu(A@W + bias) )           (N = DFF)
// mode 1: Out[b, idx[row], :] = A@W + bias (f32) (N = D)
__global__ __launch_bounds__(256) void gemm_kernel(const unsigned short* __restrict__ A,
                                                   const unsigned short* __restrict__ W,
                                                   const float* __restrict__ bias,
                                                   unsigned short* __restrict__ Hout,
                                                   float* __restrict__ Out,
                                                   const int* __restrict__ idx,
                                                   int N, int K, int mode) {
    __shared__ unsigned short As[2][128][40];   // [m][k], 128x32 (+8 pad)
    __shared__ unsigned short BsT[2][64][40];   // [n][k] transposed, 64x32 (+8 pad)

    const int m0 = blockIdx.y * 128;
    const int n0 = blockIdx.x * 64;
    const int wave = threadIdx.x >> 5;
    const int lane = threadIdx.x & 31;
    const int wm = (wave >> 1) * 32;         // wave row offset in tile
    const int wn = (wave & 1) * 32;          // wave col offset in tile
    const int half = lane >> 4;              // 0 or 1
    const int ml = lane & 15;
    const int kb = half * 8;                 // K base for this lane half

    // staging coordinates (A: two 8-elem segments / thread; B: one / thread)
    const int ar0 = threadIdx.x >> 2;               // 0..63
    const int ac0 = (threadIdx.x & 3) * 8;
    const int ar1 = (threadIdx.x + 256) >> 2;       // 64..127
    const int ac1 = ((threadIdx.x + 256) & 3) * 8;
    const int br  = threadIdx.x >> 3;               // 0..31 (k)
    const int bc8 = (threadIdx.x & 7) * 8;          // 0..56 (n)

    v8f acc[2][2];
    #pragma unroll
    for (int i = 0; i < 2; i++)
        #pragma unroll
        for (int j = 0; j < 2; j++)
            acc[i][j] = (v8f){0.f, 0.f, 0.f, 0.f, 0.f, 0.f, 0.f, 0.f};

    // prologue: stage K-chunk 0 into buffer 0 (register path)
    {
        *(uint4*)&As[0][ar0][ac0] = *(const uint4*)&A[(size_t)(m0 + ar0) * K + ac0];
        *(uint4*)&As[0][ar1][ac1] = *(const uint4*)&A[(size_t)(m0 + ar1) * K + ac1];
        uint4 wv = *(const uint4*)&W[(size_t)br * N + n0 + bc8];
        const unsigned short* ws16 = (const unsigned short*)&wv;
        #pragma unroll
        for (int q = 0; q < 8; q++) BsT[0][bc8 + q][br] = ws16[q];
    }

    const int nk = K / 32;
    for (int it = 0; it < nk; ++it) {
        const int cur = it & 1, nxt = cur ^ 1;
        const int k1 = (it + 1) * 32;
        const bool have_next = k1 < K;

        __syncthreads();   // buffer `cur` ready; buffer `nxt` free to overwrite

#if USE_ASYNC_LDS
        if (have_next) {   // A tile: global -> LDS without touching VGPRs
            async_copy16(&A[(size_t)(m0 + ar0) * K + k1 + ac0], &As[nxt][ar0][ac0]);
            async_copy16(&A[(size_t)(m0 + ar1) * K + k1 + ac1], &As[nxt][ar1][ac1]);
        }
#endif
        uint4 wv, av0, av1;
        if (have_next) {
            wv = *(const uint4*)&W[(size_t)(k1 + br) * N + n0 + bc8];
#if !USE_ASYNC_LDS
            av0 = *(const uint4*)&A[(size_t)(m0 + ar0) * K + k1 + ac0];
            av1 = *(const uint4*)&A[(size_t)(m0 + ar1) * K + k1 + ac1];
#endif
        }

        // Fragments from buffer `cur`: lane<16 -> K {0..7,16..23};
        // lane>=16 -> K {8..15,24..31}; each half = contiguous 16B -> ds_load_b128.
        v16bf afr[2], bfr[2];
        #pragma unroll
        for (int i = 0; i < 2; i++) {
            const unsigned short* p = &As[cur][wm + i * 16 + ml][kb];
            v8bf lo = *(const v8bf*)p;
            v8bf hi = *(const v8bf*)(p + 16);
            afr[i] = __builtin_shufflevector(lo, hi, 0, 1, 2, 3, 4, 5, 6, 7,
                                             8, 9, 10, 11, 12, 13, 14, 15);
        }
        #pragma unroll
        for (int j = 0; j < 2; j++) {
            const unsigned short* p = &BsT[cur][wn + j * 16 + ml][kb];
            v8bf lo = *(const v8bf*)p;
            v8bf hi = *(const v8bf*)(p + 16);
            bfr[j] = __builtin_shufflevector(lo, hi, 0, 1, 2, 3, 4, 5, 6, 7,
                                             8, 9, 10, 11, 12, 13, 14, 15);
        }
        #pragma unroll
        for (int i = 0; i < 2; i++)
            #pragma unroll
            for (int j = 0; j < 2; j++)
                acc[i][j] = __builtin_amdgcn_wmma_f32_16x16x32_bf16(
                    false, afr[i], false, bfr[j], (short)0, acc[i][j], false, false);

        if (have_next) {   // drain staged data into buffer `nxt`
            const unsigned short* ws16 = (const unsigned short*)&wv;
            #pragma unroll
            for (int q = 0; q < 8; q++) BsT[nxt][bc8 + q][br] = ws16[q];
#if !USE_ASYNC_LDS
            *(uint4*)&As[nxt][ar0][ac0] = av0;
            *(uint4*)&As[nxt][ar1][ac1] = av1;
#else
            __builtin_amdgcn_s_wait_asynccnt(0);
#endif
        }
    }

    // epilogue; C/D layout: VGPR r -> (M = r + half*8, N = ml) within 16x16 tile
    #pragma unroll
    for (int i = 0; i < 2; i++) {
        #pragma unroll
        for (int j = 0; j < 2; j++) {
            int nc = n0 + wn + j * 16 + ml;
            float bv = bias[nc];
            #pragma unroll
            for (int r = 0; r < 8; r++) {
                int mrow = m0 + wm + i * 16 + half * 8 + r;
                float v = acc[i][j][r] + bv;
                if (mode == 0) {
                    float g = 0.5f * v * (1.0f + erff(v * 0.70710678118f));
                    Hout[(size_t)mrow * N + nc] = f2bf(g);
                } else {
                    int b   = mrow >> 12;          // / CAP_
                    int tok = idx[mrow];
                    Out[((size_t)b * S_ + tok) * D_ + nc] = v;
                }
            }
        }
    }
}

static inline size_t align256(size_t v) { return (v + 255) & ~(size_t)255; }

extern "C" void kernel_launch(void* const* d_in, const int* in_sizes, int n_in,
                              void* d_out, int out_size, void* d_ws, size_t ws_size,
                              hipStream_t stream) {
    const float* x  = (const float*)d_in[0];
    const float* Wg = (const float*)d_in[1];
    const float* bg = (const float*)d_in[2];
    const float* W1 = (const float*)d_in[3];
    const float* b1 = (const float*)d_in[4];
    const float* W2 = (const float*)d_in[5];
    const float* b2 = (const float*)d_in[6];
    float* out = (float*)d_out;

    char* ws = (char*)d_ws;
    float* logits        = (float*)ws;          ws += align256((size_t)B_ * S_ * 4);
    int* idx             = (int*)ws;            ws += align256((size_t)B_ * CAP_ * 4);
    unsigned short* selA = (unsigned short*)ws; ws += align256((size_t)M_ * D_ * 2);
    unsigned short* w1b  = (unsigned short*)ws; ws += align256((size_t)D_ * DFF_ * 2);
    unsigned short* w2b  = (unsigned short*)ws; ws += align256((size_t)DFF_ * D_ * 2);
    unsigned short* h    = (unsigned short*)ws; ws += align256((size_t)M_ * DFF_ * 2);

    // 1) router logits (B*S tokens, 8 waves/block)
    router_kernel<<<(B_ * S_) / 8, 256, 0, stream>>>(x, Wg, bg, logits);
    // 2) per-batch top-CAP via bitonic sort
    topk_kernel<<<B_, 1024, 0, stream>>>(logits, idx);
    // 3) gather + convert activations / weights to bf16
    gather_kernel<<<M_, 256, 0, stream>>>(x, idx, (unsigned int*)selA);
    conv_bf16_kernel<<<(D_ * DFF_ / 4 + 255) / 256, 256, 0, stream>>>(
        (const float4*)W1, (unsigned int*)w1b, D_ * DFF_ / 4);
    conv_bf16_kernel<<<(DFF_ * D_ / 4 + 255) / 256, 256, 0, stream>>>(
        (const float4*)W2, (unsigned int*)w2b, DFF_ * D_ / 4);
    // 4) pass-through
    copy_kernel<<<(B_ * S_ * D_ / 4 + 255) / 256, 256, 0, stream>>>(
        (const float4*)x, (float4*)out, B_ * S_ * D_ / 4);
    // 5) GEMM1: h = gelu(selA @ W1 + b1)   [M_ x DFF_], K = D_
    gemm_kernel<<<dim3(DFF_ / 64, M_ / 128), 256, 0, stream>>>(
        selA, w1b, b1, h, nullptr, nullptr, DFF_, D_, 0);
    // 6) GEMM2 + scatter: out[b, idx, :] = h @ W2 + b2   [M_ x D_], K = DFF_
    gemm_kernel<<<dim3(D_ / 64, M_ / 128), 256, 0, stream>>>(
        h, w2b, b2, nullptr, out, idx, D_, DFF_, 1);
}
